// MyPointConv_2508260901520
// MI455X (gfx1250) — matrix operand order repaired and checked by reading, not compiled, not run on hardware
//
#include <hip/hip_runtime.h>

// CDNA5 / gfx1250 PointConv: edge-MLP via V_WMMA_F32_16X16X32_F16,
// segment-max via native global_atomic_max_u32 (values >= 0 after ReLU),
// output GEMM via WMMA.

typedef __attribute__((ext_vector_type(16))) _Float16 v16h;
typedef __attribute__((ext_vector_type(8)))  _Float16 v8h;
typedef __attribute__((ext_vector_type(8)))  float    v8f;
typedef __attribute__((ext_vector_type(4)))  float    v4f;
typedef __attribute__((ext_vector_type(4)))  int      v4i;

__device__ __forceinline__ v8f wmma_f16(v16h a, v16h b, v8f c) {
  // D = A(16x32 f16) * B(32x16 f16) + C(16x16 f32)
  return __builtin_amdgcn_wmma_f32_16x16x32_f16(false, a, false, b, (short)0, c,
                                                false, false);
}

__global__ void k_zero_f32(float* __restrict__ p, int n) {
  int i = blockIdx.x * blockDim.x + threadIdx.x;
  if (i < n) p[i] = 0.0f;
}

// ---------------------------------------------------------------------------
// Edge MLP: per 16-message tile:
//   msg[16x35] -> relu(msg@W1+b1)[16x64] -> relu(.@W2+b2)[16x64]
//   -> atomic max into agg[dst]
// One wave per tile, grid-stride over tiles. Weights live in registers as
// WMMA B-fragments (loaded once per wave, L2-resident).
// ---------------------------------------------------------------------------
__global__ __launch_bounds__(256) void k_edge_mlp(
    const float* __restrict__ x, const float* __restrict__ pos,
    const long long* __restrict__ ei, long long E, int Nn,
    const float* __restrict__ W1, const float* __restrict__ b1,
    const float* __restrict__ W2, const float* __restrict__ b2,
    float* __restrict__ agg, int nTiles) {
  // Per-wave LDS: h tile (16 rows x 64 f16, row stride 72 halves = 144B so
  // every v8h access is 16B aligned), plus 16 dst indices.
  __shared__ __align__(16) _Float16 hbuf[8][16 * 72];
  __shared__ __align__(16) int dstbuf[8][16];

  const int lane = threadIdx.x & 31;
  const int wid  = threadIdx.x >> 5;
  const int half = lane >> 4;   // 0: lanes 0-15, 1: lanes 16-31
  const int lp   = lane & 15;

  // ---- Preload W1 (35x64, zero-padded K to 64) as B-fragments -------------
  // B layout: lane holds column N = nt*16+lp, elements 0..7 -> K = kc*32 + half*8 + e; 8..15 -> K = kc*32 + half*8 + 16 + (e-8).
  v16h w1f[2][4];
#pragma unroll
  for (int kc = 0; kc < 2; ++kc)
#pragma unroll
    for (int nt = 0; nt < 4; ++nt) {
      const int Ncol = nt * 16 + lp;
      const int kb = kc * 32 + half * 8;
#pragma unroll
      for (int e = 0; e < 16; ++e) {
        const int k = kb + e + (e & 8);
        w1f[kc][nt][e] = (k < 35) ? (_Float16)W1[k * 64 + Ncol] : (_Float16)0.0f;
      }
    }
  // ---- Preload W2 (64x64) as B-fragments ----------------------------------
  v16h w2f[2][4];
#pragma unroll
  for (int kc = 0; kc < 2; ++kc)
#pragma unroll
    for (int nt = 0; nt < 4; ++nt) {
      const int Ncol = nt * 16 + lp;
      const int kb = kc * 32 + half * 8;
#pragma unroll
      for (int e = 0; e < 16; ++e)
        w2f[kc][nt][e] = (_Float16)W2[(kb + e + (e & 8)) * 64 + Ncol];
    }
  float b1v[4], b2v[4];
#pragma unroll
  for (int nt = 0; nt < 4; ++nt) {
    b1v[nt] = b1[nt * 16 + lp];
    b2v[nt] = b2[nt * 16 + lp];
  }

  _Float16* hrow = &hbuf[wid][0];
  int* dloc = &dstbuf[wid][0];

  const int waveGlobal = blockIdx.x * 8 + wid;
  const int waveCount  = gridDim.x * 8;

  for (int t = waveGlobal; t < nTiles; t += waveCount) {
    // Row handled by this lane (lanes L and L+16 mirror the same row).
    long long m = (long long)t * 16 + lp;
    long long srcR, dstR;
    if (m < E) {
      srcR = ei[m];
      dstR = ei[E + m];
    } else {  // appended self-loops (i,i); pad tiles with node-0 self loop
      long long s = m - E;
      if (s >= (long long)Nn) s = 0;  // duplicate self-loop: harmless under max
      srcR = s;
      dstR = s;
    }
    if (half == 0) dloc[lp] = (int)dstR;

    // ---- A-fragment, K-chunk 0: x features (K=0..31) ----------------------
    // A layout: elements 0..7 -> K = half*8 + 0..7; 8..15 -> K = half*8 + 16..23.
    const float* xr = x + srcR * 32;
    const v4f q0 = *(const v4f*)(xr + half * 8 + 0);
    const v4f q1 = *(const v4f*)(xr + half * 8 + 4);
    const v4f q2 = *(const v4f*)(xr + half * 8 + 16);
    const v4f q3 = *(const v4f*)(xr + half * 8 + 20);
    v16h a0;
#pragma unroll
    for (int e = 0; e < 4; ++e) {
      a0[e]      = (_Float16)q0[e];
      a0[4 + e]  = (_Float16)q1[e];
      a0[8 + e]  = (_Float16)q2[e];
      a0[12 + e] = (_Float16)q3[e];
    }
    // ---- A-fragment, K-chunk 1: dpos at K=32..34, rest zero ---------------
    v16h a1;
#pragma unroll
    for (int e = 0; e < 16; ++e) a1[e] = (_Float16)0.0f;
    if (half == 0) {
      const float dx = pos[srcR * 3 + 0] - pos[dstR * 3 + 0];
      const float dy = pos[srcR * 3 + 1] - pos[dstR * 3 + 1];
      const float dz = pos[srcR * 3 + 2] - pos[dstR * 3 + 2];
      a1[0] = (_Float16)dx;
      a1[1] = (_Float16)dy;
      a1[2] = (_Float16)dz;
    }

    // ---- Layer 1: 8x WMMA -------------------------------------------------
    v8f acc[4];
#pragma unroll
    for (int nt = 0; nt < 4; ++nt) {
#pragma unroll
      for (int r = 0; r < 8; ++r) acc[nt][r] = b1v[nt];
      acc[nt] = wmma_f16(a0, w1f[0][nt], acc[nt]);
      acc[nt] = wmma_f16(a1, w1f[1][nt], acc[nt]);
    }

    // ---- ReLU + transpose through LDS (C-layout -> A-layout) --------------
    // C layout: VGPR r holds (M = r + half*8, N = nt*16+lp).
#pragma unroll
    for (int nt = 0; nt < 4; ++nt) {
      const int Ncol = nt * 16 + lp;
#pragma unroll
      for (int r = 0; r < 8; ++r) {
        float v = acc[nt][r];
        v = v > 0.0f ? v : 0.0f;
        hrow[(r + half * 8) * 72 + Ncol] = (_Float16)v;
      }
    }
    // Same-wave LDS ops complete in order; fence the split DS counter.
    asm volatile("s_wait_dscnt 0" ::: "memory");

    // ---- Layer 2 A-fragments from LDS (16B-aligned v8h loads) -------------
    v16h a2[2];
#pragma unroll
    for (int kc = 0; kc < 2; ++kc) {
      const int kb = kc * 32 + half * 8;
      const v8h lo = *(const v8h*)&hrow[lp * 72 + kb];
      const v8h hi = *(const v8h*)&hrow[lp * 72 + kb + 16];
#pragma unroll
      for (int e = 0; e < 8; ++e) {
        a2[kc][e]     = lo[e];
        a2[kc][8 + e] = hi[e];
      }
    }

    // ---- Layer 2: 8x WMMA -------------------------------------------------
    v8f acc2[4];
#pragma unroll
    for (int nt = 0; nt < 4; ++nt) {
#pragma unroll
      for (int r = 0; r < 8; ++r) acc2[nt][r] = b2v[nt];
      acc2[nt] = wmma_f16(a2[0], w2f[0][nt], acc2[nt]);
      acc2[nt] = wmma_f16(a2[1], w2f[1][nt], acc2[nt]);
    }

    // ---- segment max: ReLU output >= 0 so u32-bit atomicMax == float max --
    const v4i d0 = *(const v4i*)&dloc[half * 8 + 0];
    const v4i d1 = *(const v4i*)&dloc[half * 8 + 4];
#pragma unroll
    for (int nt = 0; nt < 4; ++nt) {
      const int Ncol = nt * 16 + lp;
#pragma unroll
      for (int r = 0; r < 8; ++r) {
        float v = acc2[nt][r];
        v = v > 0.0f ? v : 0.0f;
        const int dn = (r < 4) ? d0[r] : d1[r - 4];
        unsigned int* ap = (unsigned int*)(agg + (long long)dn * 64 + Ncol);
        atomicMax(ap, __float_as_uint(v));  // -> global_atomic_max_u32
      }
    }
  }
}

// ---------------------------------------------------------------------------
// out[N,128] = agg[N,64] @ Wg + bg.  One wave per 16-row tile, 16 WMMA each.
// ---------------------------------------------------------------------------
__global__ __launch_bounds__(256) void k_out_gemm(
    const float* __restrict__ agg, const float* __restrict__ Wg,
    const float* __restrict__ bg, float* __restrict__ out, int Nn) {
  const int lane = threadIdx.x & 31;
  const int wid  = threadIdx.x >> 5;
  const int half = lane >> 4;
  const int lp   = lane & 15;

  // Preload Wg (64x128) as 2x8 B-fragments.
  v16h wgf[2][8];
#pragma unroll
  for (int kc = 0; kc < 2; ++kc)
#pragma unroll
    for (int nt = 0; nt < 8; ++nt) {
      const int Ncol = nt * 16 + lp;
      const int kb = kc * 32 + half * 8;
#pragma unroll
      for (int e = 0; e < 16; ++e)
        wgf[kc][nt][e] = (_Float16)Wg[(kb + e + (e & 8)) * 128 + Ncol];
    }
  float bgv[8];
#pragma unroll
  for (int nt = 0; nt < 8; ++nt) bgv[nt] = bg[nt * 16 + lp];

  const int rowTiles = (Nn + 15) / 16;
  const int waveGlobal = blockIdx.x * 8 + wid;
  const int waveCount  = gridDim.x * 8;

  for (int t = waveGlobal; t < rowTiles; t += waveCount) {
    int row = t * 16 + lp;
    if (row >= Nn) row = Nn - 1;  // clamp (N divisible by 16, but be safe)
    const float* ar = agg + (long long)row * 64;

    v16h af[2];
#pragma unroll
    for (int kc = 0; kc < 2; ++kc) {
      const int kb = kc * 32 + half * 8;
      const v4f q0 = *(const v4f*)(ar + kb + 0);
      const v4f q1 = *(const v4f*)(ar + kb + 4);
      const v4f q2 = *(const v4f*)(ar + kb + 16);
      const v4f q3 = *(const v4f*)(ar + kb + 20);
#pragma unroll
      for (int e = 0; e < 4; ++e) {
        af[kc][e]      = (_Float16)q0[e];
        af[kc][4 + e]  = (_Float16)q1[e];
        af[kc][8 + e]  = (_Float16)q2[e];
        af[kc][12 + e] = (_Float16)q3[e];
      }
    }

    v8f acc[8];
#pragma unroll
    for (int nt = 0; nt < 8; ++nt) {
#pragma unroll
      for (int r = 0; r < 8; ++r) acc[nt][r] = bgv[nt];
      acc[nt] = wmma_f16(af[0], wgf[0][nt], acc[nt]);
      acc[nt] = wmma_f16(af[1], wgf[1][nt], acc[nt]);
    }

#pragma unroll
    for (int nt = 0; nt < 8; ++nt) {
      const int Ncol = nt * 16 + lp;
#pragma unroll
      for (int r = 0; r < 8; ++r) {
        const int M = t * 16 + r + half * 8;
        if (M < Nn) out[(long long)M * 128 + Ncol] = acc[nt][r];
      }
    }
  }
}

extern "C" void kernel_launch(void* const* d_in, const int* in_sizes, int n_in,
                              void* d_out, int out_size, void* d_ws,
                              size_t ws_size, hipStream_t stream) {
  (void)n_in; (void)out_size; (void)ws_size;
  const float*     x   = (const float*)d_in[0];
  const float*     pos = (const float*)d_in[1];
  const long long* ei  = (const long long*)d_in[2];  // int64 [2, E]
  const float*     W1  = (const float*)d_in[3];
  const float*     b1  = (const float*)d_in[4];
  const float*     W2  = (const float*)d_in[5];
  const float*     b2  = (const float*)d_in[6];
  const float*     Wg  = (const float*)d_in[7];
  const float*     bg  = (const float*)d_in[8];

  const int       Nn = in_sizes[0] / 32;        // 50000
  const long long E  = in_sizes[2] / 2;         // 1600000

  float* agg = (float*)d_ws;                    // [Nn, 64] f32 = 12.8 MB
  const int aggN = Nn * 64;

  // 1) zero agg (ReLU outputs >= 0 and self-loops make every segment nonempty)
  k_zero_f32<<<(aggN + 255) / 256, 256, 0, stream>>>(agg, aggN);

  // 2) edge MLP + atomic-max aggregation
  const long long total = E + (long long)Nn;    // 1,650,000 (multiple of 16)
  const int nTiles = (int)((total + 15) / 16);
  k_edge_mlp<<<1024, 256, 0, stream>>>(x, pos, ei, E, Nn, W1, b1, W2, b2, agg,
                                       nTiles);

  // 3) output GEMM
  const int rowTiles = (Nn + 15) / 16;
  int blocksC = (rowTiles + 7) / 8;
  if (blocksC > 1024) blocksC = 1024;
  k_out_gemm<<<blocksC, 256, 0, stream>>>(agg, Wg, bg, (float*)d_out, Nn);
}
